// GroupedVectorAttention_35347580846874
// MI455X (gfx1250) — matrix operand order repaired and checked by reading, not compiled, not run on hardware
//
#include <hip/hip_runtime.h>
#include <hip/hip_bf16.h>

typedef __bf16 v16bf __attribute__((ext_vector_type(16)));
typedef float  v8f   __attribute__((ext_vector_type(8)));

#define N_PTS 20000
#define KNN 16
#define CCH 256
#define GRP 16
#define EPSF 1e-5f
#define MTILES (N_PTS / 16)   // 1250
#define PTS 4                 // points per attention block (B-fragment reuse factor)

// ---------------------------------------------------------------------------
// Pack kernels: fp32 -> bf16, pre-swizzled into WMMA fragment order.
// A-frag (16x32 bf16): lane l (m=l&15, hi=l>>4), elem e: K = (e>>3)*16 + hi*8 + (e&7)
// B-frag (32x16 bf16): lane l (n=l&15, hi=l>>4), elem e: K = (e>>3)*16 + hi*8 + (e&7)
// ---------------------------------------------------------------------------
__global__ void pack_feat_kernel(const float* __restrict__ feat, __bf16* __restrict__ dst) {
    int d = blockIdx.x * blockDim.x + threadIdx.x;       // < N_PTS*CCH
    int e = d & 15, l = (d >> 4) & 31, j = (d >> 9) & 7, mt = d >> 12;
    int row = mt * 16 + (l & 15);
    int k   = j * 32 + ((e >> 3) << 4) + ((l >> 4) << 3) + (e & 7);
    dst[d] = (__bf16)feat[row * CCH + k];
}

__global__ void pack_w_kernel(const float* __restrict__ src, __bf16* __restrict__ dst, int ncols) {
    int d = blockIdx.x * blockDim.x + threadIdx.x;       // 8 * (ncols/16) * 512
    int e = d & 15, l = (d >> 4) & 31, j = (d >> 9) & 7, w = d >> 12;
    int k = j * 32 + ((e >> 3) << 4) + ((l >> 4) << 3) + (e & 7);
    int n = w * 16 + (l & 15);
    dst[d] = (__bf16)src[k * ncols + n];
}

// ---------------------------------------------------------------------------
// q/k/v projection GEMM: one wave = 2x2 grid of 16x16 output tiles (operand
// reuse: each A/B fragment feeds two WMMAs -> half the L2 traffic).
// D = feat(32x256) @ W(256x32) + bias, then BN+ReLU for q,k.
// ---------------------------------------------------------------------------
__global__ __launch_bounds__(256)
void qkv_gemm_kernel(const __bf16* __restrict__ featP,
                     const __bf16* __restrict__ WqP, const __bf16* __restrict__ WkP,
                     const __bf16* __restrict__ WvP,
                     const float* __restrict__ bq, const float* __restrict__ bk,
                     const float* __restrict__ bv,
                     const float* __restrict__ bnq, const float* __restrict__ bnk,
                     float* __restrict__ qO, float* __restrict__ kO, float* __restrict__ vO) {
    int wave = threadIdx.x >> 5, lane = threadIdx.x & 31;
    int wid = blockIdx.x * 8 + wave;          // < 3 * 625 * 8 = 15000
    int nt2  = wid & 7;                       // ntile pair
    int rest = wid >> 3;
    int mt2  = rest % (MTILES / 2);           // mtile pair
    int mat  = rest / (MTILES / 2);           // 0=q 1=k 2=v
    const __bf16* WP  = (mat == 0) ? WqP : (mat == 1) ? WkP : WvP;
    const float* bias = (mat == 0) ? bq  : (mat == 1) ? bk  : bv;
    float* dst        = (mat == 0) ? qO  : (mat == 1) ? kO  : vO;

    int laneN = lane & 15, hi = lane >> 4;
    int col[2], mtile[2], ntile[2];
    v8f acc[2][2];
#pragma unroll
    for (int ni = 0; ni < 2; ni++) {
        ntile[ni] = nt2 * 2 + ni;
        col[ni] = ntile[ni] * 16 + laneN;
        float b0 = bias[col[ni]];
#pragma unroll
        for (int mi = 0; mi < 2; mi++)
#pragma unroll
            for (int e = 0; e < 8; e++) acc[mi][ni][e] = b0;
    }
    mtile[0] = mt2 * 2; mtile[1] = mt2 * 2 + 1;

#pragma unroll
    for (int j = 0; j < 8; j++) {
        v16bf a[2], b[2];
#pragma unroll
        for (int mi = 0; mi < 2; mi++) {
            const __bf16* ap = featP + ((size_t)(mtile[mi] * 8 + j) * 32 + lane) * 16;
#pragma unroll
            for (int e = 0; e < 16; e++) a[mi][e] = ap[e];
        }
#pragma unroll
        for (int ni = 0; ni < 2; ni++) {
            const __bf16* bp = WP + ((size_t)(ntile[ni] * 8 + j) * 32 + lane) * 16;
#pragma unroll
            for (int e = 0; e < 16; e++) b[ni][e] = bp[e];
        }
#pragma unroll
        for (int mi = 0; mi < 2; mi++)
#pragma unroll
            for (int ni = 0; ni < 2; ni++)
                acc[mi][ni] = __builtin_amdgcn_wmma_f32_16x16x32_bf16(
                    false, a[mi], false, b[ni], (short)0, acc[mi][ni], false, false);
    }

#pragma unroll
    for (int ni = 0; ni < 2; ni++) {
        float gmm = 0.f, bet = 0.f, mu = 0.f, vr = 0.f;
        if (mat < 2) {
            const float* bn = (mat == 0) ? bnq : bnk;
            gmm = bn[col[ni]]; bet = bn[CCH + col[ni]];
            mu = bn[2 * CCH + col[ni]]; vr = bn[3 * CCH + col[ni]];
        }
#pragma unroll
        for (int mi = 0; mi < 2; mi++) {
#pragma unroll
            for (int e = 0; e < 8; e++) {
                int row = mtile[mi] * 16 + hi * 8 + e;
                float x = acc[mi][ni][e];
                if (mat < 2) {
                    x = (x - mu) * rsqrtf(vr + EPSF) * gmm + bet;
                    x = fmaxf(x, 0.f);
                }
                dst[(size_t)row * CCH + col[ni]] = x;
            }
        }
    }
}

// ---------------------------------------------------------------------------
// Fused grouped vector attention: 1 block (16 waves) per PTS=4 points.
// Rows of every tile = the K=16 neighbors (one WMMA M-tile per point).
// Wave w owns channel columns [16w,16w+16) == group w; each Wp2 B-fragment is
// loaded once per k-step and reused across the 4 points' A-fragments.
// ---------------------------------------------------------------------------
__global__ __launch_bounds__(512)
void attn_kernel(const float* __restrict__ coord, const int* __restrict__ knn,
                 const float* __restrict__ qA, const float* __restrict__ kA,
                 const float* __restrict__ vA,
                 const float* __restrict__ Wp1, const float* __restrict__ bp1,
                 const float* __restrict__ bnp,
                 const __bf16* __restrict__ Wp2P, const float* __restrict__ bp2,
                 const __bf16* __restrict__ Ww1P, const float* __restrict__ bw1,
                 const float* __restrict__ bnw,
                 const float* __restrict__ Ww2, const float* __restrict__ bw2,
                 float* __restrict__ out) {
    __shared__ __bf16 hA[PTS][16 * 264];    // h = relu(bn(pos@Wp1+bp1)) per point
    __shared__ __bf16 relA[PTS][16 * 264];  // relation_qk in bf16 for the Ww1 GEMM
    __shared__ float posL[PTS][16][3];
    __shared__ float maskL[PTS][16];
    __shared__ int   idxL[PTS][16];
    __shared__ float preL[PTS][16 * 17];
    __shared__ float wL[PTS][16 * 17];

    int n0 = blockIdx.x * PTS;
    int tid = threadIdx.x, lane = tid & 31, wave = tid >> 5;
    int laneN = lane & 15, hi = lane >> 4;

    // --- A: neighbor indices, masks, relative positions (64 threads) ---
    if (tid < PTS * 16) {
        int p = tid >> 4, i = tid & 15;
        int n = n0 + p;
        int raw = knn[n * KNN + i];
        float m = (raw >= 0) ? 1.f : 0.f;       // sign(idx+1), idx=-1 -> 0
        int safe = raw < 0 ? 0 : raw;
        idxL[p][i] = safe; maskL[p][i] = m;
        float cx = coord[n * 3 + 0], cy = coord[n * 3 + 1], cz = coord[n * 3 + 2];
        posL[p][i][0] = (coord[safe * 3 + 0] - cx) * m;
        posL[p][i][1] = (coord[safe * 3 + 1] - cy) * m;
        posL[p][i][2] = (coord[safe * 3 + 2] - cz) * m;
    }
    __syncthreads();

    // --- B: h = relu(bn(pos @ Wp1 + bp1)) into LDS (bf16), all points ---
    {
        int row = wave;                         // 16 waves -> 16 neighbor rows
#pragma unroll
        for (int p = 0; p < PTS; p++) {
            float px = posL[p][row][0], py = posL[p][row][1], pz = posL[p][row][2];
#pragma unroll
            for (int t = 0; t < 8; t++) {
                int c = lane * 8 + t;           // 0..255
                float x = px * Wp1[c] + py * Wp1[CCH + c] + pz * Wp1[2 * CCH + c] + bp1[c];
                x = (x - bnp[2 * CCH + c]) * rsqrtf(bnp[3 * CCH + c] + EPSF) * bnp[c]
                    + bnp[CCH + c];
                hA[p][row * 264 + c] = (__bf16)fmaxf(x, 0.f);
            }
        }
    }
    __syncthreads();

    // --- C: peb = h @ Wp2 + bp2 ; B-fragment reused across PTS points ---
    int col = wave * 16 + laneN;
    float bias2 = bp2[col];
    v8f acc[PTS];
#pragma unroll
    for (int p = 0; p < PTS; p++)
#pragma unroll
        for (int e = 0; e < 8; e++) acc[p][e] = bias2;

#pragma unroll
    for (int j = 0; j < 8; j++) {
        v16bf b;
        const __bf16* bp = Wp2P + ((size_t)(wave * 8 + j) * 32 + lane) * 16;
#pragma unroll
        for (int e = 0; e < 16; e++) b[e] = bp[e];
#pragma unroll
        for (int p = 0; p < PTS; p++) {
            v16bf a;
            const __bf16* ap = &hA[p][laneN * 264 + j * 32 + hi * 8];
#pragma unroll
            for (int e = 0; e < 8; e++) { a[e] = ap[e]; a[e + 8] = ap[e + 16]; }
            acc[p] = __builtin_amdgcn_wmma_f32_16x16x32_bf16(false, a, false, b,
                                                             (short)0, acc[p], false, false);
        }
    }

    // --- D: gather k/v (L2-resident), relation & value ---
    float val[PTS][8];
#pragma unroll
    for (int p = 0; p < PTS; p++) {
        int n = n0 + p;
        float qv = qA[(size_t)n * CCH + col];
#pragma unroll
        for (int e = 0; e < 8; e++) {
            int m = hi * 8 + e;
            int im = idxL[p][m]; float mm = maskL[p][m];
            float kg = kA[(size_t)im * CCH + col] * mm;
            float vg = vA[(size_t)im * CCH + col] * mm;
            float peb = acc[p][e];
            val[p][e] = vg + peb;
            relA[p][m * 264 + col] = (__bf16)(kg - qv + peb);
        }
    }
    __syncthreads();

    // --- E: weight = softmax(relu(bn(rel@Ww1+bw1)) @ Ww2 + bw2) * mask ---
    //     waves 0..PTS-1 each handle one point (wave-uniform branch, EXEC all-1s)
    if (wave < PTS) {
        int p = wave;
        int g = laneN;
        float wb = bw1[g];
        v8f wacc;
#pragma unroll
        for (int e = 0; e < 8; e++) wacc[e] = wb;
#pragma unroll
        for (int j = 0; j < 8; j++) {
            v16bf a, b;
            const __bf16* ap = &relA[p][laneN * 264 + j * 32 + hi * 8];
            const __bf16* bp = Ww1P + ((size_t)j * 32 + lane) * 16;
#pragma unroll
            for (int e = 0; e < 8; e++) { a[e] = ap[e]; a[e + 8] = ap[e + 16]; }
#pragma unroll
            for (int e = 0; e < 16; e++) { b[e] = bp[e]; }
            wacc = __builtin_amdgcn_wmma_f32_16x16x32_bf16(false, a, false, b,
                                                           (short)0, wacc, false, false);
        }
        float gmm = bnw[g], bet = bnw[GRP + g], mu = bnw[2 * GRP + g], vr = bnw[3 * GRP + g];
#pragma unroll
        for (int e = 0; e < 8; e++) {
            int m = hi * 8 + e;
            float x = (wacc[e] - mu) * rsqrtf(vr + EPSF) * gmm + bet;
            preL[p][m * 17 + g] = fmaxf(x, 0.f);
        }
        // tiny 16x16 second linear in VALU (in-wave LDS ordering is program order)
        float wv[8];
        float bb = bw2[g];
#pragma unroll
        for (int e = 0; e < 8; e++) {
            int m = hi * 8 + e;
            float s = bb;
#pragma unroll
            for (int gp = 0; gp < 16; gp++) s += preL[p][m * 17 + gp] * Ww2[gp * 16 + g];
            wv[e] = s;
        }
        // softmax over the 16 neighbors (8 here + 8 in partner lane l^16)
        float mx = wv[0];
#pragma unroll
        for (int e = 1; e < 8; e++) mx = fmaxf(mx, wv[e]);
        mx = fmaxf(mx, __shfl_xor(mx, 16, 32));
        float sum = 0.f;
#pragma unroll
        for (int e = 0; e < 8; e++) { wv[e] = __expf(wv[e] - mx); sum += wv[e]; }
        sum += __shfl_xor(sum, 16, 32);
        float inv = 1.f / sum;
#pragma unroll
        for (int e = 0; e < 8; e++) {
            int m = hi * 8 + e;
            wL[p][m * 17 + g] = wv[e] * inv * maskL[p][m];  // mask AFTER softmax (per ref)
        }
    }
    __syncthreads();

    // --- F: out[n,g,i] = sum_k value[k, g*16+i] * weight[k,g]  (g == wave) ---
#pragma unroll
    for (int p = 0; p < PTS; p++) {
        float partial = 0.f;
#pragma unroll
        for (int e = 0; e < 8; e++) {
            int m = hi * 8 + e;
            partial += val[p][e] * wL[p][m * 17 + wave];
        }
        partial += __shfl_xor(partial, 16, 32);
        if (hi == 0) out[(size_t)(n0 + p) * CCH + col] = partial;
    }
}

// ---------------------------------------------------------------------------
extern "C" void kernel_launch(void* const* d_in, const int* in_sizes, int n_in,
                              void* d_out, int out_size, void* d_ws, size_t ws_size,
                              hipStream_t stream) {
    const float* feat  = (const float*)d_in[0];
    const float* coord = (const float*)d_in[1];
    const int*   knn   = (const int*)d_in[2];
    const float* Wq  = (const float*)d_in[3];
    const float* bq  = (const float*)d_in[4];
    const float* bnq = (const float*)d_in[5];
    const float* Wk  = (const float*)d_in[6];
    const float* bk  = (const float*)d_in[7];
    const float* bnk = (const float*)d_in[8];
    const float* Wv  = (const float*)d_in[9];
    const float* bv  = (const float*)d_in[10];
    const float* Wp1 = (const float*)d_in[11];
    const float* bp1 = (const float*)d_in[12];
    const float* bnp = (const float*)d_in[13];
    const float* Wp2 = (const float*)d_in[14];
    const float* bp2 = (const float*)d_in[15];
    const float* Ww1 = (const float*)d_in[16];
    const float* bw1 = (const float*)d_in[17];
    const float* bnw = (const float*)d_in[18];
    const float* Ww2 = (const float*)d_in[19];
    const float* bw2 = (const float*)d_in[20];
    float* out = (float*)d_out;

    char* ws = (char*)d_ws;
    size_t o = 0;
    __bf16* featP = (__bf16*)(ws + o); o += (size_t)N_PTS * CCH * 2;
    __bf16* WqP   = (__bf16*)(ws + o); o += (size_t)CCH * CCH * 2;
    __bf16* WkP   = (__bf16*)(ws + o); o += (size_t)CCH * CCH * 2;
    __bf16* WvP   = (__bf16*)(ws + o); o += (size_t)CCH * CCH * 2;
    __bf16* Wp2P  = (__bf16*)(ws + o); o += (size_t)CCH * CCH * 2;
    __bf16* Ww1P  = (__bf16*)(ws + o); o += (size_t)CCH * GRP * 2;
    o = (o + 255) & ~(size_t)255;
    float* qA = (float*)(ws + o); o += (size_t)N_PTS * CCH * 4;
    float* kO = (float*)(ws + o); o += (size_t)N_PTS * CCH * 4;
    float* vO = (float*)(ws + o); o += (size_t)N_PTS * CCH * 4;

    pack_feat_kernel<<<(N_PTS * CCH) / 256, 256, 0, stream>>>(feat, featP);
    pack_w_kernel<<<(CCH * CCH) / 256, 256, 0, stream>>>(Wq,  WqP,  CCH);
    pack_w_kernel<<<(CCH * CCH) / 256, 256, 0, stream>>>(Wk,  WkP,  CCH);
    pack_w_kernel<<<(CCH * CCH) / 256, 256, 0, stream>>>(Wv,  WvP,  CCH);
    pack_w_kernel<<<(CCH * CCH) / 256, 256, 0, stream>>>(Wp2, Wp2P, CCH);
    pack_w_kernel<<<(CCH * GRP) / 256, 256, 0, stream>>>(Ww1, Ww1P, GRP);

    // 3 mats * 625 mtile-pairs * 8 ntile-pairs = 15000 wave-tiles / 8 waves
    qkv_gemm_kernel<<<15000 / 8, 256, 0, stream>>>(
        featP, WqP, WkP, WvP, bq, bk, bv, bnq, bnk, qA, kO, vO);

    attn_kernel<<<N_PTS / PTS, 512, 0, stream>>>(
        coord, knn, qA, kO, vO, Wp1, bp1, bnp, Wp2P, bp2,
        Ww1P, bw1, bnw, Ww2, bw2, out);
}
